// TransformerBlock_45956150067882
// MI455X (gfx1250) — matrix operand order repaired and checked by reading, not compiled, not run on hardware
//
#include <hip/hip_runtime.h>
#include <hip/hip_bf16.h>
#include <stdint.h>

// ---- problem constants (match reference) ----
#define B_   2
#define S_   1024
#define D_   2048
#define H_   32
#define HKV_ 8
#define HD_  64
#define E_   8
#define I_   1024
#define T_   (B_ * S_)   // 2048 tokens

typedef __attribute__((ext_vector_type(16))) __bf16 v16bf;
typedef __attribute__((ext_vector_type(8)))  float  v8f;
typedef uint32_t u32x4 __attribute__((ext_vector_type(4)));
typedef uint32_t u32x8 __attribute__((ext_vector_type(8)));

union FragAB {
    v16bf    v;
    uint16_t u[16];
    uint4    q[2];
};

__device__ __forceinline__ uint16_t f2bf(float f) {
    union { float f; uint32_t u; } c; c.f = f;
    uint32_t u = c.u;
    return (uint16_t)((u + 0x7FFFu + ((u >> 16) & 1u)) >> 16);  // RNE
}
__device__ __forceinline__ float bf2f(uint16_t h) {
    union { uint32_t u; float f; } c; c.u = ((uint32_t)h) << 16;
    return c.f;
}

// =====================================================================
// Weight convert + transpose: W[K x N] f32 row-major -> Wt[N x K] bf16.
// =====================================================================
__global__ void transpose_bf16(const float* __restrict__ W,
                               uint16_t* __restrict__ Wt, int K, int N) {
    size_t idx = (size_t)blockIdx.x * blockDim.x + threadIdx.x;
    size_t total = (size_t)K * N;
    if (idx >= total) return;
    int k = (int)(idx / N);
    int n = (int)(idx % N);
    Wt[(size_t)n * K + k] = f2bf(W[idx]);
}

// =====================================================================
// RMSNorm row kernel: f32 in -> bf16 out (feeds WMMA A operand).
// =====================================================================
__global__ __launch_bounds__(256)
void rmsnorm_bf16(const float* __restrict__ x, const float* __restrict__ w,
                  uint16_t* __restrict__ out, int Dd) {
    int row = blockIdx.x;
    const float* xr = x + (size_t)row * Dd;
    float ss = 0.f;
    for (int d = threadIdx.x; d < Dd; d += 256) { float v = xr[d]; ss += v * v; }
    __shared__ float red[256];
    red[threadIdx.x] = ss;
    __syncthreads();
    for (int s = 128; s > 0; s >>= 1) {
        if (threadIdx.x < (unsigned)s) red[threadIdx.x] += red[threadIdx.x + s];
        __syncthreads();
    }
    float scale = rsqrtf(red[0] / (float)Dd + 1e-5f);
    for (int d = threadIdx.x; d < Dd; d += 256)
        out[(size_t)row * Dd + d] = f2bf(xr[d] * scale * w[d]);
}

// =====================================================================
// Generic bf16 WMMA GEMM: C[MxN] = A[MxK](bf16,row-major) @ Bt[NxK](bf16).
// 8 waves / block, each wave owns a 32x32 tile (2x2 v_wmma accumulators).
// B operand is staged LDS-side by the Tensor Data Mover (TDM):
//   - per-wave 32col x 32K bf16 slice, double buffered (2 x 2KB per wave)
//   - 2-D tensor descriptor (D# groups 0/1 per CDNA5 ISA)
//     built in SGPRs, issued with inline `tensor_load_to_lds`
//   - TDM ops are in-order per wave: after issuing tile k+1,
//     `s_wait_tensorcnt 0x1` guarantees tile k is resident in LDS
// A operand uses direct global_load_b128 (reused across N via L2/WGP$).
// Fragment layouts as loaded here (wave32, A/B in contiguous K order):
//   A 16x32 bf16 : lane&15 = row, half=lane>>4 -> K = half*16+{0..15}
//   B 32x16 bf16 : lane&15 = col, K = half*16+{0..15}
//   C 16x16 f32  : lane&15 = col, row = i + 8*half for acc element i
// =====================================================================
enum { EP_STORE = 0, EP_RESID = 1, EP_ROWSCALE = 2, EP_ACCUM = 3 };

__device__ __forceinline__ void tdm_issue_b_tile(const uint16_t* gsrc,
                                                 uint32_t lds_off,
                                                 int K, int N) {
    // ---- D# group 0 (128b): count=1, lds_addr, global tile addr, type=2
    const uint64_t ga = (uint64_t)(uintptr_t)gsrc;
    u32x4 g0;
    g0[0] = 1u;                                        // count = 1 valid D#
    g0[1] = lds_off;                                   // LDS byte address
    g0[2] = (uint32_t)ga;                              // global_addr[31:0]
    g0[3] = (uint32_t)((ga >> 32) & 0x01FFFFFFu)       // global_addr[56:32]
          | 0x80000000u;                               // type = 2 ("image")
    // ---- D# group 1 (256b): data_size=2B, dims/strides, no multicast
    u32x8 g1;
    g1[0] = (1u << 16);                                // data_size = 1 -> 2 bytes
    g1[1] = ((uint32_t)K & 0xFFFFu) << 16;             // tensor_dim0[15:0]
    g1[2] = ((uint32_t)K >> 16)                        // tensor_dim0[31:16]
          | (((uint32_t)N & 0xFFFFu) << 16);           // tensor_dim1[15:0]
    g1[3] = (((uint32_t)N >> 16) & 0xFFFFu)            // tensor_dim1[31:16]
          | (32u << 16);                               // tile_dim0 = 32 (K elems)
    g1[4] = 32u;                                       // tile_dim1 = 32 rows
    g1[5] = (uint32_t)K;                               // tensor_dim0_stride[31:0]
    g1[6] = 0u;                                        // stride0[47:32], stride1 lo
    g1[7] = 0u;
    asm volatile("tensor_load_to_lds %0, %1" :: "s"(g0), "s"(g1) : "memory");
}

template <int EP>
__global__ __launch_bounds__(256)
void gemm_bf16_wmma(const uint16_t* __restrict__ A,
                    const uint16_t* __restrict__ Bt,
                    float* __restrict__ C,
                    const float* __restrict__ aux,   // resid (RESID) or rowscale (ROWSCALE)
                    int M, int N, int K) {
    const int lane = threadIdx.x & 31;
    const int wid  = threadIdx.x >> 5;
    const int bm   = blockIdx.x * 128 + (wid >> 1) * 32;
    const int bn   = blockIdx.y * 64  + (wid & 1) * 32;
    const int lh   = lane >> 4;     // half
    const int lan  = lane & 15;

    // per-wave double-buffered B tile: 32 cols x 32 K bf16 (2KB) x 2
    __shared__ __align__(16) uint16_t Bsh[8][2][32 * 32];
    uint16_t* bbase0 = &Bsh[wid][0][0];
    uint16_t* bbase1 = &Bsh[wid][1][0];
    const uint32_t ldsoff0 = (uint32_t)(uintptr_t)bbase0;
    const uint32_t ldsoff1 = (uint32_t)(uintptr_t)bbase1;
    const uint16_t* bslab = Bt + (size_t)bn * K;   // wave's 32-col slice base

    v8f acc00 = {}, acc01 = {}, acc10 = {}, acc11 = {};

    const uint16_t* arow0 = A + (size_t)(bm + lan) * K;
    const uint16_t* arow1 = arow0 + (size_t)16 * K;

    // prime the pipeline with tile k0 = 0 into buffer 0
    tdm_issue_b_tile(bslab, ldsoff0, K, N);

    for (int k0 = 0; k0 < K; k0 += 32) {
        const int buf = (k0 >> 5) & 1;
        // issue next tile into the other buffer while we consume this one
        if (k0 + 32 < K)
            tdm_issue_b_tile(bslab + (k0 + 32), buf ? ldsoff0 : ldsoff1, K, N);
        // in-order per-wave TDM: <=1 outstanding leaves only the prefetch
        if (k0 + 32 < K)
            asm volatile("s_wait_tensorcnt 0x1" ::: "memory");
        else
            asm volatile("s_wait_tensorcnt 0x0" ::: "memory");

        FragAB a0, a1, b0, b1;
        const int ka = k0 + lh * 16;
        a0.q[0] = *(const uint4*)(arow0 + ka);
        a0.q[1] = *(const uint4*)(arow0 + ka + 8);
        a1.q[0] = *(const uint4*)(arow1 + ka);
        a1.q[1] = *(const uint4*)(arow1 + ka + 8);
        if (k0 + 32 < K) __builtin_prefetch(arow0 + ka + 32, 0, 0);

        const uint16_t* bp = buf ? bbase1 : bbase0;   // [col 0..31][k 0..31]
        b0.q[0] = *(const uint4*)(bp + (size_t)lan * 32 + lh * 16);
        b0.q[1] = *(const uint4*)(bp + (size_t)lan * 32 + lh * 16 + 8);
        b1.q[0] = *(const uint4*)(bp + (size_t)(16 + lan) * 32 + lh * 16);
        b1.q[1] = *(const uint4*)(bp + (size_t)(16 + lan) * 32 + lh * 16 + 8);

        acc00 = __builtin_amdgcn_wmma_f32_16x16x32_bf16(false, a0.v, false, b0.v, (short)0, acc00, false, false);
        acc01 = __builtin_amdgcn_wmma_f32_16x16x32_bf16(false, a0.v, false, b1.v, (short)0, acc01, false, false);
        acc10 = __builtin_amdgcn_wmma_f32_16x16x32_bf16(false, a1.v, false, b0.v, (short)0, acc10, false, false);
        acc11 = __builtin_amdgcn_wmma_f32_16x16x32_bf16(false, a1.v, false, b1.v, (short)0, acc11, false, false);
    }

    v8f* accs[2][2] = { { &acc00, &acc01 }, { &acc10, &acc11 } };
#pragma unroll
    for (int im = 0; im < 2; ++im)
#pragma unroll
        for (int in = 0; in < 2; ++in) {
            const int col = bn + in * 16 + lan;
            v8f a = *accs[im][in];
#pragma unroll
            for (int i = 0; i < 8; ++i) {
                const int row = bm + im * 16 + i + 8 * lh;
                const size_t idx = (size_t)row * N + col;
                float v = a[i];
                if (EP == EP_RESID)    v += aux[idx];
                if (EP == EP_ROWSCALE) v *= aux[row];
                if (EP == EP_ACCUM)    v += C[idx];
                C[idx] = v;
            }
        }
}

// =====================================================================
// RoPE + split qkv (f32) into q_bf[B,H,S,HD], k_bf[B,HKV,S,HD],
// v transposed vT[B,HKV,HD,S] (so P@V B-fragments are contiguous).
// =====================================================================
__global__ void rope_split(const float* __restrict__ qkv,
                           const float* __restrict__ rope,
                           uint16_t* __restrict__ qb,
                           uint16_t* __restrict__ kb,
                           uint16_t* __restrict__ vbT) {
    size_t idx = (size_t)blockIdx.x * blockDim.x + threadIdx.x;
    const size_t total = (size_t)T_ * 48 * 32;   // 48 heads * 32 pairs
    if (idx >= total) return;
    int p    = (int)(idx % 32);
    int head = (int)((idx / 32) % 48);
    size_t t = idx / (32 * 48);
    int b = (int)(t / S_), s = (int)(t % S_);
    const float* src = qkv + t * 3072 + (size_t)head * 64 + p * 2;
    float x0 = src[0], x1 = src[1];
    if (head < 40) {  // rope on q and k heads
        float c  = rope[((size_t)s * 32 + p) * 2 + 0];
        float sn = rope[((size_t)s * 32 + p) * 2 + 1];
        float r0 = x0 * c - x1 * sn;
        float r1 = x1 * c + x0 * sn;
        x0 = r0; x1 = r1;
    }
    if (head < 32) {
        size_t o = (((size_t)(b * H_ + head) * S_ + s) * HD_) + p * 2;
        qb[o] = f2bf(x0); qb[o + 1] = f2bf(x1);
    } else if (head < 40) {
        int hk = head - 32;
        size_t o = (((size_t)(b * HKV_ + hk) * S_ + s) * HD_) + p * 2;
        kb[o] = f2bf(x0); kb[o + 1] = f2bf(x1);
    } else {
        int hk = head - 40;
        size_t base = (size_t)(b * HKV_ + hk) * HD_;
        vbT[(base + p * 2 + 0) * S_ + s] = f2bf(x0);
        vbT[(base + p * 2 + 1) * S_ + s] = f2bf(x1);
    }
}

// =====================================================================
// Flash attention, causal, GQA (N_REP=4). One wave (32 threads) per
// (b, h, 16-row q tile). Scores via WMMA, online softmax in C-fragment
// layout, P re-shaped through LDS into A-fragment layout, P@V via WMMA.
// =====================================================================
__global__ __launch_bounds__(32)
void attn_fa(const uint16_t* __restrict__ qb, const uint16_t* __restrict__ kbuf,
             const uint16_t* __restrict__ vbT, uint16_t* __restrict__ attn_out) {
    const int lane = threadIdx.x;
    const int lh   = lane >> 4;
    const int lan  = lane & 15;
    const int q0 = blockIdx.x * 16;
    const int h  = blockIdx.y;
    const int b  = blockIdx.z;
    const int hk = h >> 2;   // N_REP = H/HKV = 4

    const uint16_t* Q  = qb   + (size_t)(b * H_   + h ) * S_  * HD_;
    const uint16_t* Kp = kbuf + (size_t)(b * HKV_ + hk) * S_  * HD_;
    const uint16_t* Vt = vbT  + (size_t)(b * HKV_ + hk) * HD_ * S_;

    FragAB aQ[2];
    {
        const uint16_t* qrow = Q + (size_t)(q0 + lan) * HD_;
#pragma unroll
        for (int ds2 = 0; ds2 < 2; ++ds2) {
            const int ka = ds2 * 32 + lh * 16;
            aQ[ds2].q[0] = *(const uint4*)(qrow + ka);
            aQ[ds2].q[1] = *(const uint4*)(qrow + ka + 8);
        }
    }

    float m[8], lsum[8];
    v8f o[4] = {};
#pragma unroll
    for (int i = 0; i < 8; ++i) { m[i] = -3.0e38f; lsum[i] = 0.f; }

    __shared__ __align__(16) uint16_t Pld[16 * 32];

    for (int k0 = 0; k0 < q0 + 16; k0 += 32) {
        v8f sc[2] = {};
#pragma unroll
        for (int sub = 0; sub < 2; ++sub) {
            const uint16_t* kcol = Kp + (size_t)(k0 + sub * 16 + lan) * HD_;
#pragma unroll
            for (int ds2 = 0; ds2 < 2; ++ds2) {
                FragAB bK;
                const int kk = ds2 * 32 + lh * 16;
                bK.q[0] = *(const uint4*)(kcol + kk);
                bK.q[1] = *(const uint4*)(kcol + kk + 8);
                sc[sub] = __builtin_amdgcn_wmma_f32_16x16x32_bf16(
                    false, aQ[ds2].v, false, bK.v, (short)0, sc[sub], false, false);
            }
        }
        float pe0[8], pe1[8];
#pragma unroll
        for (int i = 0; i < 8; ++i) {
            const int qi = q0 + i + 8 * lh;
            float s0 = sc[0][i] * 0.125f; if (k0 + lan      > qi) s0 = -3.0e38f;
            float s1 = sc[1][i] * 0.125f; if (k0 + 16 + lan > qi) s1 = -3.0e38f;
            float tm = fmaxf(s0, s1);
#pragma unroll
            for (int d2 = 1; d2 < 16; d2 <<= 1) tm = fmaxf(tm, __shfl_xor(tm, d2, 32));
            const float mn = fmaxf(m[i], tm);
            const float alpha = __expf(m[i] - mn);
            const float e0 = __expf(s0 - mn);
            const float e1 = __expf(s1 - mn);
            float rs = e0 + e1;
#pragma unroll
            for (int d2 = 1; d2 < 16; d2 <<= 1) rs += __shfl_xor(rs, d2, 32);
            lsum[i] = lsum[i] * alpha + rs;
            m[i] = mn;
#pragma unroll
            for (int g = 0; g < 4; ++g) o[g][i] = o[g][i] * alpha;
            pe0[i] = e0; pe1[i] = e1;
        }
#pragma unroll
        for (int i = 0; i < 8; ++i) {
            Pld[(i + 8 * lh) * 32 + lan]      = f2bf(pe0[i]);
            Pld[(i + 8 * lh) * 32 + 16 + lan] = f2bf(pe1[i]);
        }
        __syncthreads();
        FragAB pF;
        {
            const uint16_t* prow = Pld + (size_t)lan * 32;
            pF.q[0] = *(const uint4*)(prow + lh * 16);
            pF.q[1] = *(const uint4*)(prow + lh * 16 + 8);
        }
#pragma unroll
        for (int g = 0; g < 4; ++g) {
            FragAB bV;
            const uint16_t* vcol = Vt + (size_t)(g * 16 + lan) * S_ + k0 + lh * 16;
            bV.q[0] = *(const uint4*)(vcol);
            bV.q[1] = *(const uint4*)(vcol + 8);
            o[g] = __builtin_amdgcn_wmma_f32_16x16x32_bf16(
                false, pF.v, false, bV.v, (short)0, o[g], false, false);
        }
        __syncthreads();
    }
#pragma unroll
    for (int i = 0; i < 8; ++i) {
        const int row = q0 + i + 8 * lh;
        const float inv = 1.f / lsum[i];
        const size_t base = ((size_t)(b * S_) + row) * (H_ * HD_) + (size_t)h * HD_;
#pragma unroll
        for (int g = 0; g < 4; ++g)
            attn_out[base + g * 16 + lan] = f2bf(o[g][i] * inv);
    }
}

// =====================================================================
// Router: logits = g @ router_w (D x 8), top-2 + sigmoid -> scores[E][T]
// =====================================================================
__global__ __launch_bounds__(256)
void router_topk(const uint16_t* __restrict__ g, const float* __restrict__ rw,
                 float* __restrict__ scores) {
    const int t = blockIdx.x;
    float acc[E_] = {};
    const uint16_t* row = g + (size_t)t * D_;
    for (int d = threadIdx.x; d < D_; d += 256) {
        const float xv = bf2f(row[d]);
#pragma unroll
        for (int e2 = 0; e2 < E_; ++e2) acc[e2] += xv * rw[(size_t)d * E_ + e2];
    }
    __shared__ float red[E_][256];
#pragma unroll
    for (int e2 = 0; e2 < E_; ++e2) red[e2][threadIdx.x] = acc[e2];
    __syncthreads();
    for (int s = 128; s > 0; s >>= 1) {
        if (threadIdx.x < (unsigned)s)
#pragma unroll
            for (int e2 = 0; e2 < E_; ++e2) red[e2][threadIdx.x] += red[e2][threadIdx.x + s];
        __syncthreads();
    }
    if (threadIdx.x == 0) {
        float lg[E_];
#pragma unroll
        for (int e2 = 0; e2 < E_; ++e2) lg[e2] = red[e2][0];
        int b1 = 0;
        for (int e2 = 1; e2 < E_; ++e2) if (lg[e2] > lg[b1]) b1 = e2;
        int b2 = -1;
        for (int e2 = 0; e2 < E_; ++e2) {
            if (e2 == b1) continue;
            if (b2 < 0 || lg[e2] > lg[b2]) b2 = e2;
        }
        for (int e2 = 0; e2 < E_; ++e2) {
            float s = 0.f;
            if (e2 == b1 || e2 == b2) s = 1.f / (1.f + __expf(-lg[e2]));
            scores[(size_t)e2 * T_ + t] = s;
        }
    }
}

// silu(gate) * up -> bf16 (feeds next WMMA GEMM's A operand)
__global__ void silu_mul_bf16(const float* __restrict__ gsrc,
                              const float* __restrict__ usrc,
                              uint16_t* __restrict__ out,
                              int cols, int gstride, int ustride) {
    size_t idx = (size_t)blockIdx.x * blockDim.x + threadIdx.x;
    if (idx >= (size_t)T_ * cols) return;
    size_t row = idx / cols;
    int c = (int)(idx % cols);
    const float gv = gsrc[row * gstride + c];
    const float uv = usrc[row * ustride + c];
    const float sg = gv / (1.f + __expf(-gv));
    out[row * cols + c] = f2bf(sg * uv);
}

__global__ void add_final(const float* __restrict__ a, const float* __restrict__ b,
                          float* __restrict__ out, size_t n) {
    size_t idx = (size_t)blockIdx.x * blockDim.x + threadIdx.x;
    if (idx < n) out[idx] = a[idx] + b[idx];
}

// =====================================================================
// kernel_launch
// =====================================================================
extern "C" void kernel_launch(void* const* d_in, const int* in_sizes, int n_in,
                              void* d_out, int out_size, void* d_ws, size_t ws_size,
                              hipStream_t stream) {
    (void)in_sizes; (void)n_in; (void)out_size; (void)ws_size;
    const float* x           = (const float*)d_in[0];
    const float* rope_cache  = (const float*)d_in[2];
    const float* w_qkv       = (const float*)d_in[4];
    const float* w_o         = (const float*)d_in[5];
    const float* norm1_w     = (const float*)d_in[6];
    const float* norm2_w     = (const float*)d_in[7];
    const float* router_w    = (const float*)d_in[8];
    const float* w_gate_up   = (const float*)d_in[9];
    const float* w_down      = (const float*)d_in[10];
    const float* shared_gate = (const float*)d_in[11];
    const float* shared_up   = (const float*)d_in[12];
    const float* shared_down = (const float*)d_in[13];
    float* out = (float*)d_out;

    char* p = (char*)d_ws;
    auto alloc = [&](size_t bytes) -> void* {
        void* r = (void*)p;
        p += (bytes + 255) & ~(size_t)255;
        return r;
    };
    uint16_t* wqkvT  = (uint16_t*)alloc((size_t)3072 * D_ * 2);
    uint16_t* woT    = (uint16_t*)alloc((size_t)D_ * (H_ * HD_) * 2);
    uint16_t* sgT    = (uint16_t*)alloc((size_t)I_ * D_ * 2);
    uint16_t* suT    = (uint16_t*)alloc((size_t)I_ * D_ * 2);
    uint16_t* sdT    = (uint16_t*)alloc((size_t)D_ * I_ * 2);
    uint16_t* wguT   = (uint16_t*)alloc((size_t)E_ * 2 * I_ * D_ * 2);
    uint16_t* wdT    = (uint16_t*)alloc((size_t)E_ * D_ * I_ * 2);
    uint16_t* h_bf   = (uint16_t*)alloc((size_t)T_ * D_ * 2);
    float*    qkv    = (float*)   alloc((size_t)T_ * 3072 * 4);
    uint16_t* q_bf   = (uint16_t*)alloc((size_t)T_ * H_ * HD_ * 2);
    uint16_t* k_bf   = (uint16_t*)alloc((size_t)T_ * HKV_ * HD_ * 2);
    uint16_t* v_bfT  = (uint16_t*)alloc((size_t)T_ * HKV_ * HD_ * 2);
    uint16_t* attn_bf= (uint16_t*)alloc((size_t)T_ * H_ * HD_ * 2);
    float*    h1     = (float*)   alloc((size_t)T_ * D_ * 4);
    uint16_t* g_bf   = (uint16_t*)alloc((size_t)T_ * D_ * 2);
    float*    scores = (float*)   alloc((size_t)E_ * T_ * 4);
    float*    tmp1   = (float*)   alloc((size_t)T_ * I_ * 4);
    float*    tmp2   = (float*)   alloc((size_t)T_ * I_ * 4);
    uint16_t* hid_sh = (uint16_t*)alloc((size_t)T_ * I_ * 2);
    float*    moe_acc= (float*)   alloc((size_t)T_ * D_ * 4);
    float*    gu     = (float*)   alloc((size_t)T_ * 2 * I_ * 4);
    uint16_t* hid_e  = (uint16_t*)alloc((size_t)T_ * I_ * 2);

    auto g1d = [](size_t n) { return dim3((unsigned)((n + 255) / 256)); };

    // ---- weight conversion (transpose + f32->bf16) ----
    transpose_bf16<<<g1d((size_t)D_ * 3072), 256, 0, stream>>>(w_qkv, wqkvT, D_, 3072);
    transpose_bf16<<<g1d((size_t)(H_ * HD_) * D_), 256, 0, stream>>>(w_o, woT, H_ * HD_, D_);
    transpose_bf16<<<g1d((size_t)D_ * I_), 256, 0, stream>>>(shared_gate, sgT, D_, I_);
    transpose_bf16<<<g1d((size_t)D_ * I_), 256, 0, stream>>>(shared_up, suT, D_, I_);
    transpose_bf16<<<g1d((size_t)I_ * D_), 256, 0, stream>>>(shared_down, sdT, I_, D_);
    for (int e = 0; e < E_; ++e) {
        transpose_bf16<<<g1d((size_t)D_ * 2 * I_), 256, 0, stream>>>(
            w_gate_up + (size_t)e * D_ * 2 * I_, wguT + (size_t)e * 2 * I_ * D_, D_, 2 * I_);
        transpose_bf16<<<g1d((size_t)I_ * D_), 256, 0, stream>>>(
            w_down + (size_t)e * I_ * D_, wdT + (size_t)e * D_ * I_, I_, D_);
    }

    // ---- attention path ----
    rmsnorm_bf16<<<T_, 256, 0, stream>>>(x, norm1_w, h_bf, D_);
    gemm_bf16_wmma<EP_STORE><<<dim3(T_ / 128, 3072 / 64), 256, 0, stream>>>(
        h_bf, wqkvT, qkv, nullptr, T_, 3072, D_);
    rope_split<<<g1d((size_t)T_ * 48 * 32), 256, 0, stream>>>(qkv, rope_cache, q_bf, k_bf, v_bfT);
    attn_fa<<<dim3(S_ / 16, H_, B_), 32, 0, stream>>>(q_bf, k_bf, v_bfT, attn_bf);
    gemm_bf16_wmma<EP_RESID><<<dim3(T_ / 128, D_ / 64), 256, 0, stream>>>(
        attn_bf, woT, h1, x, T_, D_, H_ * HD_);

    // ---- MoE path ----
    rmsnorm_bf16<<<T_, 256, 0, stream>>>(h1, norm2_w, g_bf, D_);
    router_topk<<<T_, 256, 0, stream>>>(g_bf, router_w, scores);

    gemm_bf16_wmma<EP_STORE><<<dim3(T_ / 128, I_ / 64), 256, 0, stream>>>(
        g_bf, sgT, tmp1, nullptr, T_, I_, D_);
    gemm_bf16_wmma<EP_STORE><<<dim3(T_ / 128, I_ / 64), 256, 0, stream>>>(
        g_bf, suT, tmp2, nullptr, T_, I_, D_);
    silu_mul_bf16<<<g1d((size_t)T_ * I_), 256, 0, stream>>>(tmp1, tmp2, hid_sh, I_, I_, I_);
    gemm_bf16_wmma<EP_STORE><<<dim3(T_ / 128, D_ / 64), 256, 0, stream>>>(
        hid_sh, sdT, moe_acc, nullptr, T_, D_, I_);

    for (int e = 0; e < E_; ++e) {
        gemm_bf16_wmma<EP_ROWSCALE><<<dim3(T_ / 128, (2 * I_) / 64), 256, 0, stream>>>(
            g_bf, wguT + (size_t)e * 2 * I_ * D_, gu, scores + (size_t)e * T_, T_, 2 * I_, D_);
        silu_mul_bf16<<<g1d((size_t)T_ * I_), 256, 0, stream>>>(
            gu, gu + I_, hid_e, I_, 2 * I_, 2 * I_);
        gemm_bf16_wmma<EP_ACCUM><<<dim3(T_ / 128, D_ / 64), 256, 0, stream>>>(
            hid_e, wdT + (size_t)e * D_ * I_, moe_acc, nullptr, T_, D_, I_);
    }

    add_final<<<g1d((size_t)T_ * D_), 256, 0, stream>>>(h1, moe_acc, out, (size_t)T_ * D_);
}
